// GraphAttention_59691455480307
// MI455X (gfx1250) — matrix operand order repaired and checked by reading, not compiled, not run on hardware
//
#include <hip/hip_runtime.h>
#include <hip/hip_bf16.h>

typedef __attribute__((ext_vector_type(16))) __bf16 v16bf;
typedef __attribute__((ext_vector_type(8)))  float  v8f;

#define SEQ   2048
#define INF   128
#define OUTF  64
#define BSZ   8

// ---------------------------------------------------------------------------
// K1: h = node @ W (f32), s1 = h@a[:64], s2 = h@a[64:], hT = bf16(h)^T
// ---------------------------------------------------------------------------
__global__ __launch_bounds__(128) void gat_h_kernel(
    const float* __restrict__ node, const float* __restrict__ W,
    const float* __restrict__ a, __bf16* __restrict__ hT,
    float* __restrict__ s1, float* __restrict__ s2)
{
  const int idx = blockIdx.x * blockDim.x + threadIdx.x;   // 0..16383
  const int b = idx >> 11;
  const int s = idx & (SEQ - 1);

  const float* nr = node + (size_t)idx * INF;
  float h[OUTF];
#pragma unroll
  for (int o = 0; o < OUTF; ++o) h[o] = 0.f;

  for (int k = 0; k < INF; ++k) {
    const float nk = nr[k];
    const float* wr = W + k * OUTF;
#pragma unroll
    for (int o = 0; o < OUTF; ++o) h[o] = fmaf(nk, wr[o], h[o]);
  }

  float t1 = 0.f, t2 = 0.f;
  __bf16* hp = hT + (size_t)b * OUTF * SEQ + s;
#pragma unroll
  for (int o = 0; o < OUTF; ++o) {
    t1 = fmaf(h[o], a[o], t1);
    t2 = fmaf(h[o], a[OUTF + o], t2);
    hp[(size_t)o * SEQ] = (__bf16)h[o];
  }
  s1[idx] = t1;
  s2[idx] = t2;
}

// ---------------------------------------------------------------------------
// K2: s2max[b] = max_j s2[b][j]  (safe static softmax max per row)
// ---------------------------------------------------------------------------
__global__ __launch_bounds__(256) void gat_s2max_kernel(
    const float* __restrict__ s2, float* __restrict__ s2m)
{
  __shared__ float red[256];
  const int b = blockIdx.x;
  const int tid = threadIdx.x;
  float m = -3.4e38f;
  for (int j = tid; j < SEQ; j += 256) m = fmaxf(m, s2[b * SEQ + j]);
  red[tid] = m;
  __syncthreads();
  for (int off = 128; off > 0; off >>= 1) {
    if (tid < off) red[tid] = fmaxf(red[tid], red[tid + off]);
    __syncthreads();
  }
  if (tid == 0) s2m[b] = red[0];
}

// ---------------------------------------------------------------------------
// K3: fused masked-softmax attention + P@h via v_wmma_f32_16x16x32_bf16.
// Block = 4 waves, one 16-row i-tile; wave w streams j in [w*512,(w+1)*512).
// Static max m_i = lrelu(s1[i] + max(s2)); all math branchless so the
// compiler cannot predicate/sink loads or the exp.
// ---------------------------------------------------------------------------
__global__ __launch_bounds__(128) void gat_attn_kernel(
    const float* __restrict__ adj, const __bf16* __restrict__ hT,
    const float* __restrict__ s1, const float* __restrict__ s2,
    const float* __restrict__ s2max, float* __restrict__ out)
{
  __shared__ float accL[4][16][64];
  __shared__ float lsumL[4][32];

  const int b    = blockIdx.y;
  const int i0   = blockIdx.x << 4;
  const int lane = threadIdx.x & 31;
  const int w    = threadIdx.x >> 5;
  const int r    = lane & 15;        // A row / B column within tile
  const int hi   = lane >> 4;        // half-wave select

  const float s1r = s1[b * SEQ + i0 + r];
  const float xm  = s1r + s2max[b];
  const float mr  = fmaxf(xm, 0.01f * xm);   // leakyrelu (exact, branchless)

  v8f acc[4] = {};
  float lsum0 = 0.f, lsum1 = 0.f;

  const float*  adjrow = adj + (((size_t)b * SEQ) + i0 + r) * SEQ;
  const float*  s2row  = s2 + b * SEQ;
  const __bf16* hTb    = hT + (size_t)b * OUTF * SEQ;
  const int koffA = hi << 4;    // A-side K base: lanes 0-15 -> 0..15, 16-31 -> 16..31
  const int koffB = hi << 4;    // B-side K base: lanes 0-15 -> 0..15, 16-31 -> 16..31

  for (int c = 0; c < 16; ++c) {
    const int j0 = (w << 9) + (c << 5);

    // adj + s2 in this kernel's A k order (k = 16*hi + t): one contiguous 16-float run
    float4 a0 = *(const float4*)(adjrow + j0 + koffA);
    float4 a1 = *(const float4*)(adjrow + j0 + koffA + 4);
    float4 a2 = *(const float4*)(adjrow + j0 + koffA + 8);
    float4 a3 = *(const float4*)(adjrow + j0 + koffA + 12);
    float4 q0 = *(const float4*)(s2row + j0 + koffA);
    float4 q1 = *(const float4*)(s2row + j0 + koffA + 4);
    float4 q2 = *(const float4*)(s2row + j0 + koffA + 8);
    float4 q3 = *(const float4*)(s2row + j0 + koffA + 12);

    float av[16] = {a0.x,a0.y,a0.z,a0.w, a1.x,a1.y,a1.z,a1.w,
                    a2.x,a2.y,a2.z,a2.w, a3.x,a3.y,a3.z,a3.w};
    float sv[16] = {q0.x,q0.y,q0.z,q0.w, q1.x,q1.y,q1.z,q1.w,
                    q2.x,q2.y,q2.z,q2.w, q3.x,q3.y,q3.z,q3.w};

    v16bf afrag = {};
#pragma unroll
    for (int t = 0; t < 16; ++t) {
      // branchless: exp always computed, mask folded in as a multiply
      const float e   = fmaxf(s1r + sv[t], 0.01f * (s1r + sv[t]));  // leakyrelu
      const float msk = (__builtin_fabsf(av[t]) > 1e-6f) ? 1.0f : 0.0f;
      const float p   = __expf(e - mr) * msk;
      if (t & 1) lsum1 += p; else lsum0 += p;
      afrag[t] = (__bf16)p;
    }

    // B operand: hT[n][j0+koffB .. +15], contiguous 32B per lane
    const __bf16* hp = hTb + (size_t)r * SEQ + j0 + koffB;
    v16bf b0 = *(const v16bf*)(hp);
    v16bf b1 = *(const v16bf*)(hp + (size_t)16 * SEQ);
    v16bf b2 = *(const v16bf*)(hp + (size_t)32 * SEQ);
    v16bf b3 = *(const v16bf*)(hp + (size_t)48 * SEQ);

    acc[0] = __builtin_amdgcn_wmma_f32_16x16x32_bf16(false, afrag, false, b0, (short)0, acc[0], false, false);
    acc[1] = __builtin_amdgcn_wmma_f32_16x16x32_bf16(false, afrag, false, b1, (short)0, acc[1], false, false);
    acc[2] = __builtin_amdgcn_wmma_f32_16x16x32_bf16(false, afrag, false, b2, (short)0, acc[2], false, false);
    acc[3] = __builtin_amdgcn_wmma_f32_16x16x32_bf16(false, afrag, false, b3, (short)0, acc[3], false, false);
  }

  // C/D layout: VGPR k -> row i = k + 8*hi; column n = tile*16 + (lane&15)
#pragma unroll
  for (int t = 0; t < 4; ++t) {
#pragma unroll
    for (int k = 0; k < 8; ++k)
      accL[w][k + (hi << 3)][(t << 4) + r] = acc[t][k];
  }
  lsumL[w][lane] = lsum0 + lsum1;
  __syncthreads();

  // Combine 4 j-slices, normalize, ELU, store coalesced (128 thr x 8 elems)
#pragma unroll
  for (int e = 0; e < 8; ++e) {
    const int f = (e << 7) + threadIdx.x;
    const int i = f >> 6;
    const int n = f & 63;
    float sum = accL[0][i][n] + accL[1][i][n] + accL[2][i][n] + accL[3][i][n];
    float ls  = lsumL[0][i] + lsumL[0][i + 16] + lsumL[1][i] + lsumL[1][i + 16]
              + lsumL[2][i] + lsumL[2][i + 16] + lsumL[3][i] + lsumL[3][i + 16];
    float v = (ls > 0.f) ? sum / ls : 0.f;
    out[(((size_t)b * SEQ) + i0 + i) * OUTF + n] = v > 0.f ? v : (__expf(v) - 1.f);
  }
}

// ---------------------------------------------------------------------------
extern "C" void kernel_launch(void* const* d_in, const int* in_sizes, int n_in,
                              void* d_out, int out_size, void* d_ws, size_t ws_size,
                              hipStream_t stream) {
  const float* node = (const float*)d_in[0];   // [8,2048,128]
  const float* adj  = (const float*)d_in[1];   // [8,2048,2048]
  const float* W    = (const float*)d_in[2];   // [128,64]
  const float* a    = (const float*)d_in[3];   // [128,1]
  float* out = (float*)d_out;                  // [8,2048,64]

  // workspace: hT bf16 (2 MB) | s1 (64 KB) | s2 (64 KB) | s2max (32 B)
  __bf16* hT = (__bf16*)d_ws;
  float* s1  = (float*)((char*)d_ws + (size_t)BSZ * OUTF * SEQ * 2);
  float* s2  = s1 + BSZ * SEQ;
  float* s2m = s2 + BSZ * SEQ;

  gat_h_kernel<<<dim3((BSZ * SEQ) / 128), dim3(128), 0, stream>>>(node, W, a, hT, s1, s2);
  gat_s2max_kernel<<<dim3(BSZ), dim3(256), 0, stream>>>(s2, s2m);
  gat_attn_kernel<<<dim3(SEQ / 16, BSZ), dim3(128), 0, stream>>>(adj, hT, s1, s2, s2m, out);
}
